// Gj_12652973654181
// MI455X (gfx1250) — compile-verified, matrix operand order checked by reading, not run on hardware
//
#include <hip/hip_runtime.h>
#include <stdint.h>

typedef _Float16 v16h __attribute__((ext_vector_type(16)));
typedef _Float16 v8h  __attribute__((ext_vector_type(8)));
typedef _Float16 v4h  __attribute__((ext_vector_type(4)));
typedef float    v8f  __attribute__((ext_vector_type(8)));

#define NSPE 4
#define BM 64
#define BK 32
#define BN 512        // full NMAX per block -> each gathered rho row loads once
#define PITCH 48      // halves per LDS row: 32 data + 16 pad (96B; 16B-aligned, 24-bank stride)
#define NTHREADS 512  // 16 wave32

// ---------------------------------------------------------------------------
// Kernel 0: zero the per-species counters in workspace.
// ---------------------------------------------------------------------------
__global__ void zero_counts_kernel(int* __restrict__ counts) {
    if (threadIdx.x < NSPE) counts[threadIdx.x] = 0;
}

// ---------------------------------------------------------------------------
// Kernel 1: bucket atoms by species.  Bucket order is nondeterministic
// (atomicAdd) but every atom's output value is order-independent, so d_out
// stays deterministic.
// ---------------------------------------------------------------------------
__global__ void bucket_kernel(const int* __restrict__ symbols,
                              int* __restrict__ counts,
                              int* __restrict__ bucket,
                              int nta) {
    int i = blockIdx.x * blockDim.x + threadIdx.x;
    if (i < nta) {
        int s = symbols[i];
        int pos = atomicAdd(&counts[s], 1);
        bucket[(size_t)s * nta + pos] = i;
    }
}

// ---------------------------------------------------------------------------
// Kernel 2: one-time convert + transpose of W: Wt[spe][n][k] = (f16)W[spe][k][n].
// Tiled through LDS so both global read and write are coalesced.  2MB output
// stays L2-resident for the whole GEMM.
// ---------------------------------------------------------------------------
__global__ __launch_bounds__(256)
void wt_transpose_kernel(const float* __restrict__ W, _Float16* __restrict__ Wt,
                         int O, int NMAX) {
    __shared__ _Float16 tile[32][33];
    const int spe = blockIdx.z;
    const int nt  = blockIdx.x * 32;
    const int kt  = blockIdx.y * 32;
    const int tx  = threadIdx.x & 31;
    const int ty  = threadIdx.x >> 5;   // 0..7
    const float*    Wsp = W  + (size_t)spe * O * NMAX;
    _Float16*       Wts = Wt + (size_t)spe * NMAX * O;
    #pragma unroll
    for (int j = 0; j < 4; ++j) {
        int k = kt + ty + j * 8;
        tile[ty + j * 8][tx] = (_Float16)Wsp[(size_t)k * NMAX + nt + tx];
    }
    __syncthreads();
    #pragma unroll
    for (int j = 0; j < 4; ++j) {
        int n = nt + ty + j * 8;
        Wts[(size_t)n * O + kt + tx] = tile[tx][ty + j * 8];
    }
}

// ---------------------------------------------------------------------------
// Kernel 3: grouped gather-GEMM.  512 threads (16 wave32) compute a 64 x 512
// f32 tile for one species with v_wmma_f32_16x16x32_f16.  Each wave owns a
// 32x64 sub-tile (2x4 accumulators, 8 WMMAs per k-step).  B (pre-transposed
// f16 weights) is staged with GLOBAL_LOAD_ASYNC_TO_LDS_B128 (ASYNCcnt) so the
// DMA bypasses VGPRs and overlaps the A convert + WMMAs.
// ---------------------------------------------------------------------------
__global__ __launch_bounds__(NTHREADS)
void grouped_gemm_kernel(const float*    __restrict__ rho,    // [nta, O] f32
                         const _Float16* __restrict__ Wt,     // [NSPE, NMAX, O] f16
                         const float*    __restrict__ bias,   // [NSPE, NMAX] f32
                         const int*      __restrict__ counts,
                         const int*      __restrict__ bucket,
                         float*          __restrict__ out,    // [nta, NMAX] f32
                         int nta, int O, int NMAX) {
    const int spe   = blockIdx.y;
    const int count = counts[spe];
    const int m0    = blockIdx.x * BM;
    if (m0 >= count) return;                // block-uniform early exit

    __shared__ int      idx_lds[BM];
    __shared__ _Float16 As[BM * PITCH];
    __shared__ _Float16 Bs[BN * PITCH];

    const int t    = threadIdx.x;
    const int lane = t & 31;
    const int wave = t >> 5;                // 0..15
    const int lrow = lane & 15;             // m (or n) within a 16x16 tile
    const int lhi  = lane >> 4;             // lane group 0 / 1
    const int kb   = lhi * 8;               // per-lane K base (halves)
    const int mw   = (wave & 1) * 32;       // wave M offset
    const int nw   = (wave >> 1) * 64;      // wave N offset (8 column groups)

    if (t < BM) {
        int p  = m0 + t;
        int pc = p < count ? p : count - 1; // clamp ragged tail (stores masked)
        idx_lds[t] = bucket[(size_t)spe * nta + pc];
    }
    __syncthreads();

    // A staging: thread t -> row ar (0..63), 4 consecutive k.
    const int ar = t >> 3;
    const int ac = (t & 7) * 4;
    const size_t arow = (size_t)idx_lds[ar] * (size_t)O + ac;

    // B staging: thread t -> Bs row n = t; 64B per k-step, async DMA to LDS.
    const _Float16* wtrow = Wt + ((size_t)spe * NMAX + (size_t)t) * O;
    // LDS byte offset of this thread's Bs row (flat->LDS: low 32 bits).
    const uint32_t bs_lds = (uint32_t)(uintptr_t)(&Bs[t * PITCH]);

    v8f acc[2][4] = {};
    union F16 { v16h v; v8h h[2]; };

    const int ksteps = O / BK;
    for (int ks = 0; ks < ksteps; ++ks) {
        const int k0 = ks * BK;

        // ---- stage B: 64B global -> LDS via async DMA (no VGPR round-trip).
        //      INST_OFFSET is added to both the LDS and global addresses.
        {
            uint64_t gaddr = (uint64_t)(uintptr_t)(wtrow + k0);
            asm volatile(
                "global_load_async_to_lds_b128 %0, %1, off\n\t"
                "global_load_async_to_lds_b128 %0, %1, off offset:16\n\t"
                "global_load_async_to_lds_b128 %0, %1, off offset:32\n\t"
                "global_load_async_to_lds_b128 %0, %1, off offset:48"
                :: "v"(bs_lds), "v"(gaddr) : "memory");
        }
        // ---- stage A (gathered rho rows), f32 -> f16 ----
        {
            float4 f = *(const float4*)(rho + arow + k0);
            v4h pk;
            pk[0] = (_Float16)f.x; pk[1] = (_Float16)f.y;
            pk[2] = (_Float16)f.z; pk[3] = (_Float16)f.w;
            *(v4h*)&As[ar * PITCH + ac] = pk;
        }
        // All of this wave's async B writes must land before the barrier.
        asm volatile("s_wait_asynccnt 0x0" ::: "memory");
        __syncthreads();

        // ---- fragment loads + 8 WMMAs per wave ----
        F16 af[2], bf[4];
        #pragma unroll
        for (int im = 0; im < 2; ++im) {
            int m = mw + im * 16 + lrow;
            af[im].h[0] = *(const v8h*)&As[m * PITCH + kb];
            af[im].h[1] = *(const v8h*)&As[m * PITCH + kb + 16];
        }
        #pragma unroll
        for (int in = 0; in < 4; ++in) {
            int n = nw + in * 16 + lrow;
            bf[in].h[0] = *(const v8h*)&Bs[n * PITCH + kb];
            bf[in].h[1] = *(const v8h*)&Bs[n * PITCH + kb + 16];
        }
        #pragma unroll
        for (int im = 0; im < 2; ++im)
            #pragma unroll
            for (int in = 0; in < 4; ++in)
                acc[im][in] = __builtin_amdgcn_wmma_f32_16x16x32_f16(
                    false, af[im].v, false, bf[in].v,
                    (short)0, acc[im][in], false, false);
        __syncthreads();
    }

    // ---- epilogue: hoist per-row scatter addresses, add bias, masked store ----
    float* rptr[2][8];
    bool   rok[2][8];
    #pragma unroll
    for (int im = 0; im < 2; ++im)
        #pragma unroll
        for (int vg = 0; vg < 8; ++vg) {
            int ml = mw + im * 16 + vg + lhi * 8;       // local M row
            rok[im][vg]  = (m0 + ml) < count;
            rptr[im][vg] = out + (size_t)idx_lds[ml] * NMAX;
        }
    #pragma unroll
    for (int in = 0; in < 4; ++in) {
        const int nl = nw + in * 16 + lrow;
        const float bv = bias[(size_t)spe * NMAX + nl];
        #pragma unroll
        for (int im = 0; im < 2; ++im) {
            v8f c = acc[im][in];
            #pragma unroll
            for (int vg = 0; vg < 8; ++vg)
                if (rok[im][vg]) rptr[im][vg][nl] = c[vg] + bv;
        }
    }
}

// ---------------------------------------------------------------------------
extern "C" void kernel_launch(void* const* d_in, const int* in_sizes, int n_in,
                              void* d_out, int out_size, void* d_ws, size_t ws_size,
                              hipStream_t stream) {
    const float* rho     = (const float*)d_in[0];
    const float* W       = (const float*)d_in[1];
    const float* bias    = (const float*)d_in[2];
    const int*   symbols = (const int*)d_in[3];
    float*       out     = (float*)d_out;

    const int nta  = in_sizes[3];
    const int O    = in_sizes[0] / nta;          // 512
    const int nmax = in_sizes[2] / NSPE;         // 512 (== BN)

    // workspace layout: counts | buckets | Wt (f16, transposed)
    int*  counts = (int*)d_ws;
    int*  bucket = (int*)((char*)d_ws + 256);
    size_t wt_off = 256 + (((size_t)NSPE * nta * sizeof(int) + 255) & ~(size_t)255);
    _Float16* Wt = (_Float16*)((char*)d_ws + wt_off);

    zero_counts_kernel<<<1, 32, 0, stream>>>(counts);
    bucket_kernel<<<(nta + 255) / 256, 256, 0, stream>>>(symbols, counts, bucket, nta);

    dim3 tgrid(nmax / 32, O / 32, NSPE);
    wt_transpose_kernel<<<tgrid, 256, 0, stream>>>(W, Wt, O, nmax);

    dim3 grid((nta + BM - 1) / BM, NSPE);
    grouped_gemm_kernel<<<grid, NTHREADS, 0, stream>>>(rho, Wt, bias, counts, bucket,
                                                       out, nta, O, nmax);
}
